// TernaryExpert_10806137717331
// MI455X (gfx1250) — matrix operand order repaired and checked by reading, not compiled, not run on hardware
//
#include <hip/hip_runtime.h>
#include <math.h>

typedef __attribute__((ext_vector_type(16))) _Float16 v16h;
typedef __attribute__((ext_vector_type(8)))  float    v8f;
typedef __attribute__((ext_vector_type(4)))  int      v4i;

#define TERN_EPS 1e-8f

#if __has_builtin(__builtin_amdgcn_global_load_async_to_lds_b128) && \
    __has_builtin(__builtin_amdgcn_s_wait_asynccnt)
#define HAS_ASYNC_LDS 1
#else
#define HAS_ASYNC_LDS 0
#endif

// Builtin signature (from clang diagnostic): (v4i as1* src, v4i as3* dst, Ii, Ii)
// generic -> global (as1): VA-identity.  generic -> LDS (as3): low 32 bits of a
// generic LDS pointer are the LDS byte offset (ISA 10.2).
#define ASG(p) ((v4i __attribute__((address_space(1)))*)(uintptr_t)(p))
#define ASL(p) ((v4i __attribute__((address_space(3)))*)(uint32_t)(uintptr_t)(p))

union AV32 {
    float4   f4[2];
    v16h     v;
    _Float16 h[16];
};

// ---------------------------------------------------------------------------
// Zero-init a float buffer (needed each launch: hSumSq + absmean accumulators)
// ---------------------------------------------------------------------------
__global__ __launch_bounds__(256) void init_zero_kernel(float* __restrict__ p, int n) {
    int i = blockIdx.x * blockDim.x + threadIdx.x;
    if (i < n) p[i] = 0.0f;
}

// ---------------------------------------------------------------------------
// Per-row RMS norm of x (f32, d=1024) -> f16.  One 256-thread block per row.
// ---------------------------------------------------------------------------
__global__ __launch_bounds__(256) void rmsnorm_f32_to_f16(const float* __restrict__ x,
                                                          _Float16* __restrict__ xn,
                                                          int d) {
    __shared__ float red[8];
    __shared__ float s_scale;
    const int row = blockIdx.x;
    const int t   = threadIdx.x;

    const float4 v = ((const float4*)(x + (size_t)row * d))[t];
    float ss = v.x * v.x + v.y * v.y + v.z * v.z + v.w * v.w;
    for (int m = 16; m > 0; m >>= 1) ss += __shfl_xor(ss, m, 32);
    if ((t & 31) == 0) red[t >> 5] = ss;
    __syncthreads();
    if (t == 0) {
        float s = 0.0f;
        for (int i = 0; i < 8; ++i) s += red[i];
        float rms = sqrtf(s) / sqrtf((float)d);
        s_scale = 1.0f / (rms + TERN_EPS);
    }
    __syncthreads();
    const float sc = s_scale;

    union { _Float16 h[4]; float2 f; } o;
    o.h[0] = (_Float16)(v.x * sc);
    o.h[1] = (_Float16)(v.y * sc);
    o.h[2] = (_Float16)(v.z * sc);
    o.h[3] = (_Float16)(v.w * sc);
    ((float2*)(xn + (size_t)row * d))[t] = o.f;
}

// ---------------------------------------------------------------------------
// Sum of |w| over n elements -> atomic accumulate into *acc
// ---------------------------------------------------------------------------
__global__ __launch_bounds__(256) void abssum_kernel(const float* __restrict__ w, int n,
                                                     float* __restrict__ acc) {
    __shared__ float red[8];
    float s = 0.0f;
    for (int i = blockIdx.x * blockDim.x + threadIdx.x; i < n; i += gridDim.x * blockDim.x)
        s += fabsf(w[i]);
    for (int m = 16; m > 0; m >>= 1) s += __shfl_xor(s, m, 32);
    if ((threadIdx.x & 31) == 0) red[threadIdx.x >> 5] = s;
    __syncthreads();
    if (threadIdx.x == 0) {
        float t = 0.0f;
        for (int i = 0; i < 8; ++i) t += red[i];
        atomicAdd(acc, t);
    }
}

// ---------------------------------------------------------------------------
// Ternarize: wq = sign(w) * (|w| > 0.5 * mean|w|), as f16 (exactly {-1,0,1}).
// ---------------------------------------------------------------------------
__global__ __launch_bounds__(256) void ternarize_kernel(const float* __restrict__ w,
                                                        _Float16* __restrict__ wq, int n,
                                                        const float* __restrict__ absSum) {
    const float thr = 0.5f * (*absSum / (float)n);
    for (int i = blockIdx.x * blockDim.x + threadIdx.x; i < n; i += gridDim.x * blockDim.x) {
        float v = w[i];
        float t = (fabsf(v) > thr) ? (v > 0.0f ? 1.0f : -1.0f) : 0.0f;
        wq[i] = (_Float16)t;
    }
}

// ---------------------------------------------------------------------------
// Tiled WMMA GEMM:  D[m][n] = sum_k A[m][k] * B[n][k]   (B stored [N][K])
//   block tile 128x128x32, 8 waves, wave tile 32x64 (2x4 16x16 fragments).
//   Double-buffered LDS.  Preferred pipeline: GLOBAL_LOAD_ASYNC_TO_LDS_B128
//   (ASYNCcnt) stages tile i+1 with zero VGPR footprint while WMMAs run on
//   tile i; one barrier per k-step.  Fallback: register staging with the
//   prefetch live only across the WMMA block.
// ---------------------------------------------------------------------------
template <bool FUSE_GELU>
__global__ __launch_bounds__(256) void gemm_wmma_kernel(
    const _Float16* __restrict__ A,       // [M][K] f16
    const _Float16* __restrict__ B,       // [N][K] f16 (ternary weights)
    const float* __restrict__ rowSumSq,   // used if !FUSE_GELU
    float rsqrtD,                         // 1/sqrt(K) for rms
    _Float16* __restrict__ Hout,          // FUSE_GELU
    float* __restrict__ hSumSq,           // FUSE_GELU
    float* __restrict__ Fout,             // !FUSE_GELU
    int K, int N) {
    constexpr int LDT  = 40;        // padded LDS row stride (80B): 16B aligned,
                                    // 20-bank shift/row -> conflict-free b128 reads
    constexpr int BUFH = 128 * LDT; // halves per buffer
    __shared__ _Float16 Alds[2 * BUFH];
    __shared__ _Float16 Blds[2 * BUFH];

    const int tid   = threadIdx.x;
    const int lane  = tid & 31;
    const int wave  = tid >> 5;
    const int waveM = wave & 3;   // 0..3 -> 32 rows each
    const int waveN = wave >> 2;  // 0..1 -> 64 cols each
    const int mBlock = blockIdx.y * 128;
    const int nBlock = blockIdx.x * 128;

    // cooperative staging: 256 threads, each moves 16 halves (32B) per tile
    const int ldRow = tid >> 1;         // 0..127
    const int ldOff = (tid & 1) * 16;   // 0 | 16 halves within the 32-wide K slab

    v8f acc[2][4];
#pragma unroll
    for (int mi = 0; mi < 2; ++mi)
#pragma unroll
        for (int ni = 0; ni < 4; ++ni)
#pragma unroll
            for (int e = 0; e < 8; ++e) acc[mi][ni][e] = 0.0f;

    const _Float16* gA = A + (size_t)(mBlock + ldRow) * K + ldOff;
    const _Float16* gB = B + (size_t)(nBlock + ldRow) * K + ldOff;
    const int slot = ldRow * LDT + ldOff;

    // fragment addressing (wave32, 16-bit A 16x32 / B 32x16): lane half h holds k = 16h..16h+15
    const int aFragRow  = (waveM * 32 + (lane & 15)) * LDT;
    const int kSelA     = (lane < 16) ? 0 : 16;                      // halves
    const int bFragBase = (waveN * 64 + (lane & 15)) * LDT + ((lane < 16) ? 0 : 16);

    auto loadFrags = [&](int buf, v16h (&af)[2], v16h (&bf)[4]) {
        const _Float16* Ab = &Alds[buf * BUFH];
        const _Float16* Bb = &Blds[buf * BUFH];
#pragma unroll
        for (int mi = 0; mi < 2; ++mi) {
            const _Float16* pa = Ab + aFragRow + mi * 16 * LDT + kSelA;
            AV32 t;
            t.f4[0] = *(const float4*)(pa);        // K = kSel .. kSel+7
            t.f4[1] = *(const float4*)(pa + 8);    // K = kSel+8 .. kSel+15
            af[mi] = t.v;
        }
#pragma unroll
        for (int ni = 0; ni < 4; ++ni) {
            const _Float16* pb = Bb + bFragBase + ni * 16 * LDT;
            AV32 t;
            t.f4[0] = *(const float4*)(pb);        // K = base .. base+7
            t.f4[1] = *(const float4*)(pb + 8);    // K = base+8 .. base+15
            bf[ni] = t.v;
        }
    };
    auto runWmma = [&](const v16h (&af)[2], const v16h (&bf)[4]) {
#pragma unroll
        for (int mi = 0; mi < 2; ++mi)
#pragma unroll
            for (int ni = 0; ni < 4; ++ni)
                acc[mi][ni] = __builtin_amdgcn_wmma_f32_16x16x32_f16(
                    false, af[mi], false, bf[ni], (short)0, acc[mi][ni], false, false);
    };

    const int nk = K >> 5;  // K/32; 32 (gemm1) or 128 (gemm2) -> always even

#if HAS_ASYNC_LDS
    // -------- hardware async global->LDS pipeline (zero VGPR staging) ------
    auto asyncStage = [&](int buf, int k0) {
        const _Float16* sa = gA + k0;
        const _Float16* sb = gB + k0;
        _Float16* da = &Alds[buf * BUFH + slot];
        _Float16* db = &Blds[buf * BUFH + slot];
        __builtin_amdgcn_global_load_async_to_lds_b128(ASG(sa),     ASL(da),     0, 0);
        __builtin_amdgcn_global_load_async_to_lds_b128(ASG(sa + 8), ASL(da + 8), 0, 0);
        __builtin_amdgcn_global_load_async_to_lds_b128(ASG(sb),     ASL(db),     0, 0);
        __builtin_amdgcn_global_load_async_to_lds_b128(ASG(sb + 8), ASL(db + 8), 0, 0);
    };
    asyncStage(0, 0);
    for (int i = 0; i < nk; i += 2) {
        __builtin_amdgcn_s_wait_asynccnt(0);   // own tile-i copies landed
        __syncthreads();                       // everyone's landed; buf1 free
        if (i + 1 < nk) asyncStage(1, (i + 1) << 5);  // runs during WMMAs
        {
            v16h af[2], bf[4];
            loadFrags(0, af, bf);
            runWmma(af, bf);
        }
        __builtin_amdgcn_s_wait_asynccnt(0);
        __syncthreads();
        if (i + 2 < nk) asyncStage(0, (i + 2) << 5);
        {
            v16h af[2], bf[4];
            loadFrags(1, af, bf);
            runWmma(af, bf);
        }
    }
#else
    // -------- register-staged fallback: prefetch lives only across WMMAs ---
    auto loadTile = [&](int k0, AV32& a, AV32& b) {
        a.f4[0] = *(const float4*)(gA + k0);
        a.f4[1] = *(const float4*)(gA + k0 + 8);
        b.f4[0] = *(const float4*)(gB + k0);
        b.f4[1] = *(const float4*)(gB + k0 + 8);
    };
    auto stageTile = [&](int buf, const AV32& a, const AV32& b) {
        _Float16* as = &Alds[buf * BUFH + slot];
        _Float16* bs = &Blds[buf * BUFH + slot];
        *(float4*)(as)     = a.f4[0];
        *(float4*)(as + 8) = a.f4[1];
        *(float4*)(bs)     = b.f4[0];
        *(float4*)(bs + 8) = b.f4[1];
    };
    auto step = [&](int buf, int nbuf, int k0, bool hasNext) {
        v16h af[2], bf[4];
        loadFrags(buf, af, bf);            // ds_loads issue first
        AV32 a, b;
        if (hasNext) loadTile(k0, a, b);   // global loads overlap ds + WMMAs
        runWmma(af, bf);
        if (hasNext) stageTile(nbuf, a, b);
    };
    {
        AV32 a, b;
        loadTile(0, a, b);
        stageTile(0, a, b);
    }
    for (int i = 0; i < nk; i += 2) {
        __syncthreads();
        step(0, 1, (i + 1) << 5, (i + 1) < nk);
        __syncthreads();
        step(1, 0, (i + 2) << 5, (i + 2) < nk);
    }
#endif

    // --------------------------- epilogue ---------------------------------
    const int rOff = (lane >> 4) * 8;  // C/D: lanes 16..31 hold rows M+8..M+15
    const int c16  = lane & 15;
    if (FUSE_GELU) {
#pragma unroll
        for (int mi = 0; mi < 2; ++mi) {
            float ss[8];
#pragma unroll
            for (int j = 0; j < 8; ++j) ss[j] = 0.0f;
            const int rowBase = mBlock + waveM * 32 + mi * 16 + rOff;
#pragma unroll
            for (int ni = 0; ni < 4; ++ni) {
                const int col = nBlock + waveN * 64 + ni * 16 + c16;
#pragma unroll
                for (int j = 0; j < 8; ++j) {
                    float v = acc[mi][ni][j];
                    float g = 0.5f * v * (1.0f + erff(v * 0.70710678118654752f));
                    Hout[(size_t)(rowBase + j) * N + col] = (_Float16)g;
                    ss[j] += g * g;
                }
            }
#pragma unroll
            for (int j = 0; j < 8; ++j) {  // reduce over the 16 cols in this half-wave
                float s = ss[j];
                s += __shfl_xor(s, 1, 32);
                s += __shfl_xor(s, 2, 32);
                s += __shfl_xor(s, 4, 32);
                s += __shfl_xor(s, 8, 32);
                if (c16 == 0) atomicAdd(&hSumSq[rowBase + j], s);
            }
        }
    } else {
        // per-row RMS scale commutes with the K-sum: apply once to the result
#pragma unroll
        for (int mi = 0; mi < 2; ++mi) {
            const int rowBase = mBlock + waveM * 32 + mi * 16 + rOff;
            float sclv[8];
#pragma unroll
            for (int j = 0; j < 8; ++j) {
                float ssj = rowSumSq[rowBase + j];
                sclv[j] = 1.0f / (sqrtf(ssj) * rsqrtD + TERN_EPS);
            }
#pragma unroll
            for (int ni = 0; ni < 4; ++ni) {
                const int col = nBlock + waveN * 64 + ni * 16 + c16;
#pragma unroll
                for (int j = 0; j < 8; ++j)
                    Fout[(size_t)(rowBase + j) * N + col] = acc[mi][ni][j] * sclv[j];
            }
        }
    }
}

// ---------------------------------------------------------------------------
// Launch: x[4,4096,1024]f32, w_up[4096,1024]f32, w_down[1024,4096]f32
//   -> out[4,4096,1024]f32
// ---------------------------------------------------------------------------
extern "C" void kernel_launch(void* const* d_in, const int* in_sizes, int n_in,
                              void* d_out, int out_size, void* d_ws, size_t ws_size,
                              hipStream_t stream) {
    const float* x      = (const float*)d_in[0];
    const float* w_up   = (const float*)d_in[1];
    const float* w_down = (const float*)d_in[2];
    float* out = (float*)d_out;

    const int Mtok = 16384;  // 4 * 4096 tokens
    const int Dm   = 1024;
    const int Dff  = 4096;

    // workspace layout (all 256B-aligned)
    char* ws = (char*)d_ws;
    _Float16* xn    = (_Float16*)(ws);                    //  32 MB  normalized x (f16)
    _Float16* wq_up = (_Float16*)(ws + 33554432ull);      //   8 MB  ternary w_up (f16)
    _Float16* wq_dn = (_Float16*)(ws + 41943040ull);      //   8 MB  ternary w_down (f16)
    _Float16* h     = (_Float16*)(ws + 50331648ull);      // 128 MB  gelu output (f16)
    float*    hss   = (float*)(ws + 184549376ull);        //  64 KB  per-row sum(h^2)
    float*    accs  = (float*)(ws + 184614912ull);        //   8 B   |w| sums (contig w/ hss)

    // 1) zero hss (16384) + accs (2) — contiguous
    init_zero_kernel<<<(16386 + 255) / 256, 256, 0, stream>>>(hss, 16386);

    // 2) rms-normalize x -> f16
    rmsnorm_f32_to_f16<<<Mtok, 256, 0, stream>>>(x, xn, Dm);

    // 3) mean|w| reductions
    abssum_kernel<<<512, 256, 0, stream>>>(w_up,   Dff * Dm, accs + 0);
    abssum_kernel<<<512, 256, 0, stream>>>(w_down, Dm * Dff, accs + 1);

    // 4) ternarize weights to f16 (layout [out_feat][in_feat] == GEMM B[N][K])
    ternarize_kernel<<<1024, 256, 0, stream>>>(w_up,   wq_up, Dff * Dm, accs + 0);
    ternarize_kernel<<<1024, 256, 0, stream>>>(w_down, wq_dn, Dm * Dff, accs + 1);

    // 5) GEMM1: h = gelu(xn @ w_up^T), fused row-sumsq of gelu output
    dim3 g1(Dff / 128, Mtok / 128);  // (32, 128)
    gemm_wmma_kernel<true><<<g1, 256, 0, stream>>>(xn, wq_up, nullptr, 0.0f,
                                                   h, hss, nullptr, Dm, Dff);

    // 6) GEMM2: out = rmsnorm(h) @ w_down^T  (scale applied in epilogue)
    dim3 g2(Dm / 128, Mtok / 128);   // (8, 128)
    gemm_wmma_kernel<false><<<g2, 256, 0, stream>>>(h, wq_dn, hss, 1.0f / 64.0f,
                                                    nullptr, nullptr, out, Dff, Dm);
}
